// ConvLSM_25881472926531
// MI455X (gfx1250) — matrix operand order repaired and checked by reading, not compiled, not run on hardware
//
#include <hip/hip_runtime.h>
#include <hip/hip_bf16.h>

typedef __attribute__((ext_vector_type(16))) _Float16 v16h;
typedef __attribute__((ext_vector_type(8)))  float    v8f;

union AFrag { v16h v; unsigned u[8]; _Float16 h[16]; };

#define T_STEPS 24
#define BATCH   32
#define HDIM    32
#define WDIM    32
#define RCH     64
#define NPIX    (BATCH*HDIM*WDIM)       // 32768
#define ALPHA   0.9f
#define BETA    0.9f

// ---------------------------------------------------------------------------
// Weight repack: OIHW f32 -> f16 WMMA B-fragments.
// Fragment layout (per CDNA5 ISA 16-bit B 32x16): lane holds column N=lane&15,
// K = (lane&16)/2 + (j&7) + 2*(j&8) for j=0..15 -> one 32B load per lane.
// frag index = ((nblk*ks2 + kpos)*chunks + kc), 512 halves per fragment.
// ---------------------------------------------------------------------------
__global__ void pack_weights_kernel(const float* __restrict__ w,
                                    _Float16* __restrict__ pk,
                                    int cout, int cin, int ks2) {
    int idx = blockIdx.x * blockDim.x + threadIdx.x;
    int total = cout * cin * ks2;
    if (idx >= total) return;
    int j    = idx & 15;
    int lane = (idx >> 4) & 31;
    int frag = idx >> 9;
    int chunks = cin >> 5;
    int kc   = frag % chunks;
    int t2   = frag / chunks;
    int kpos = t2 % ks2;
    int nblk = t2 / ks2;
    int n    = (nblk << 4) + (lane & 15);
    int chan = (kc << 5) + ((lane & 16) >> 1) + (j & 7) + ((j & 8) << 1);
    pk[idx] = (_Float16)w[(n * cin + chan) * ks2 + kpos];
}

// ---------------------------------------------------------------------------
// Input conv (C=2, 5x5, pad 2), direct VALU, channels-last output [B,H,W,R].
// Lanes within a wave share the pixel -> activation loads are wave-uniform.
// ---------------------------------------------------------------------------
__global__ __launch_bounds__(256)
void input_cur_kernel(const float* __restrict__ xt,     // [B,2,H,W]
                      const float* __restrict__ w_in,   // [R,2,5,5]
                      const float* __restrict__ b_in,   // [R]
                      float* __restrict__ cur) {        // [B,H,W,R]
    int idx = blockIdx.x * blockDim.x + threadIdx.x;    // NPIX*RCH threads
    const int r  = idx & 63;
    int pix      = idx >> 6;
    const int xx = pix & 31;
    const int yy = (pix >> 5) & 31;
    const int b  = pix >> 10;
    float s = b_in[r];
    #pragma unroll
    for (int c = 0; c < 2; ++c) {
        const float* xb = xt + (b * 2 + c) * 1024;
        const float* wb = w_in + (r * 2 + c) * 25;
        #pragma unroll
        for (int ky = 0; ky < 5; ++ky) {
            int ys = yy + ky - 2;
            if (ys < 0 || ys >= HDIM) continue;
            #pragma unroll
            for (int kx = 0; kx < 5; ++kx) {
                int xs = xx + kx - 2;
                if (xs < 0 || xs >= WDIM) continue;
                s += xb[ys * WDIM + xs] * wb[ky * 5 + kx];
            }
        }
    }
    cur[idx] = s;
}

// ---------------------------------------------------------------------------
// One LIF timestep: rec = conv5x5(spk_in, w_rec) via implicit-GEMM WMMA,
// fused synaptic/membrane update + spike emission in the epilogue.
// Wave tile = 16 pixels (one half-row) x 16 output channels; K = 25*64.
// Tile coordinates are readfirstlane'd so the kpos row check is a scalar
// branch (EXEC stays all-1s for WMMA, per ISA requirement) and tile address
// math lives in SALU.
// ---------------------------------------------------------------------------
__global__ __launch_bounds__(256)
void rsnn_step_kernel(const _Float16* __restrict__ spk_in,  // [B,H,W,R] f16
                      const float* __restrict__ cur,        // [B,H,W,R]
                      const _Float16* __restrict__ wrec_pk, // packed fragments
                      const float* __restrict__ b_rec,      // [R]
                      float* __restrict__ syn,              // [B,H,W,R]
                      float* __restrict__ mem,              // [B,H,W,R]
                      _Float16* __restrict__ spk_out,       // [B,H,W,R] f16
                      float* __restrict__ spk_rec_out,      // [B,R,H,W] f32 (d_out)
                      float* __restrict__ psum) {           // partition sum [B,H,W,R]
    const int lane = threadIdx.x & 31;
    const int wave = __builtin_amdgcn_readfirstlane(
        blockIdx.x * (blockDim.x >> 5) + (threadIdx.x >> 5));
    const int rblk = wave & 3;
    int pr = wave >> 2;
    const int x0 = (pr & 1) << 4;  pr >>= 1;
    const int y  = pr & 31;
    const int b  = pr >> 5;

    const int m      = lane & 15;
    const int xcol   = x0 + m;
    const int laneKB = (lane & 16) >> 1;   // A-fragment half-wave K offset

    v8f acc = {};
    for (int kpos = 0; kpos < 25; ++kpos) {
        const int dy = kpos / 5 - 2;
        const int dx = kpos % 5 - 2;
        const int ys = y + dy;                     // scalar (uniform)
        if (ys < 0 || ys >= HDIM) continue;        // scalar branch, EXEC untouched
        const int xs = xcol + dx;
        const bool vx = (xs >= 0) && (xs < WDIM);  // per-lane (per-M-row) validity
        const _Float16* abase =
            spk_in + (((b * HDIM + ys) * WDIM + (vx ? xs : 0)) * RCH) + laneKB;
        #pragma unroll
        for (int kc = 0; kc < 2; ++kc) {
            AFrag a;
            const _Float16* ap = abase + kc * 32;
            #pragma unroll
            for (int p = 0; p < 8; ++p) {
                unsigned v = *(const unsigned*)(ap + ((p & 3) * 2 + ((p & 4) << 2)));
                a.u[p] = vx ? v : 0u;
            }
            const v16h bf = *(const v16h*)(wrec_pk +
                (((rblk * 25 + kpos) * 2 + kc) << 9) + (lane << 4));
            acc = __builtin_amdgcn_wmma_f32_16x16x32_f16(
                false, a.v, false, bf, (short)0, acc, false, false);
        }
    }

    // Fused LIF epilogue (D element (m,n): vgpr g -> m = g + (lane&16)/2, n = lane&15)
    const int n0 = rblk << 4;
    #pragma unroll
    for (int g = 0; g < 8; ++g) {
        const int mg  = g + ((lane & 16) >> 1);
        const int n   = lane & 15;
        const int r   = n0 + n;
        const int xg  = x0 + mg;
        const int pix = (b * HDIM + y) * WDIM + xg;
        const int ci  = pix * RCH + r;
        float rec   = acc[g] + b_rec[r];
        float s_old = syn[ci];
        float m_old = mem[ci];
        float reset = (m_old - 1.0f > 0.0f) ? 1.0f : 0.0f;  // from PREVIOUS mem
        float s_new = ALPHA * s_old + cur[ci] + rec;
        float m_new = BETA * m_old + s_new - reset;         // THRESHOLD = 1
        float sp    = (m_new - 1.0f > 0.0f) ? 1.0f : 0.0f;
        syn[ci] = s_new;
        mem[ci] = m_new;
        spk_out[ci] = (_Float16)sp;
        spk_rec_out[((b * RCH + r) * HDIM + y) * WDIM + xg] = sp;
        psum[ci] += sp;
    }
}

// ---------------------------------------------------------------------------
// Partition means -> channels-last f16 readout [B,H,W,192] (concat order p*64+r)
// ---------------------------------------------------------------------------
__global__ void build_readout_kernel(const float* __restrict__ psum,  // [3][NPIX*64]
                                     _Float16* __restrict__ ro) {     // [NPIX*192]
    int idx = blockIdx.x * blockDim.x + threadIdx.x;
    int cg  = idx % 192;
    int pix = idx / 192;
    int p = cg / 64, r = cg % 64;
    ro[idx] = (_Float16)(psum[p * (NPIX * RCH) + pix * RCH + r] * 0.125f);
}

// ---------------------------------------------------------------------------
// Generic 3x3 WMMA conv (pad=KS/2), ReLU, channels-last in/out.
// ---------------------------------------------------------------------------
template <int CIN, int COUT, int KS, bool OUT_F16>
__global__ __launch_bounds__(256)
void conv_wmma_kernel(const _Float16* __restrict__ act,  // [B,H,W,CIN]
                      const _Float16* __restrict__ wpk,
                      const float* __restrict__ bias,
                      _Float16* __restrict__ out16,
                      float* __restrict__ out32) {
    constexpr int PAD = KS / 2;
    constexpr int CHUNKS = CIN / 32;
    constexpr int NBLK = COUT / 16;
    const int lane = threadIdx.x & 31;
    const int wave = __builtin_amdgcn_readfirstlane(
        blockIdx.x * (blockDim.x >> 5) + (threadIdx.x >> 5));
    const int nblk = wave % NBLK;
    int pr = wave / NBLK;
    const int x0 = (pr & 1) << 4;  pr >>= 1;
    const int y  = pr & 31;
    const int b  = pr >> 5;
    const int m      = lane & 15;
    const int xcol   = x0 + m;
    const int laneKB = (lane & 16) >> 1;

    v8f acc = {};
    for (int kpos = 0; kpos < KS * KS; ++kpos) {
        const int dy = kpos / KS - PAD;
        const int dx = kpos % KS - PAD;
        const int ys = y + dy;                     // scalar (uniform)
        if (ys < 0 || ys >= HDIM) continue;        // scalar branch
        const int xs = xcol + dx;
        const bool vx = (xs >= 0) && (xs < WDIM);
        const _Float16* abase =
            act + (((b * HDIM + ys) * WDIM + (vx ? xs : 0)) * CIN) + laneKB;
        #pragma unroll 2
        for (int kc = 0; kc < CHUNKS; ++kc) {
            AFrag a;
            const _Float16* ap = abase + kc * 32;
            #pragma unroll
            for (int p = 0; p < 8; ++p) {
                unsigned v = *(const unsigned*)(ap + ((p & 3) * 2 + ((p & 4) << 2)));
                a.u[p] = vx ? v : 0u;
            }
            const v16h bf = *(const v16h*)(wpk +
                (((nblk * KS * KS + kpos) * CHUNKS + kc) << 9) + (lane << 4));
            acc = __builtin_amdgcn_wmma_f32_16x16x32_f16(
                false, a.v, false, bf, (short)0, acc, false, false);
        }
    }
    #pragma unroll
    for (int g = 0; g < 8; ++g) {
        const int mg  = g + ((lane & 16) >> 1);
        const int cch = (nblk << 4) + (lane & 15);
        const int pix = (b * HDIM + y) * WDIM + x0 + mg;
        float v = acc[g] + bias[cch];
        v = v > 0.0f ? v : 0.0f;
        if (OUT_F16) out16[pix * COUT + cch] = (_Float16)v;
        else         out32[pix * COUT + cch] = v;
    }
}

// ---------------------------------------------------------------------------
// FC (131072 -> 10, NCHW flatten order) + softmax. One block per batch row.
// ---------------------------------------------------------------------------
__global__ __launch_bounds__(256)
void fc_softmax_kernel(const float* __restrict__ h2,    // [B*H*W,128] channels-last
                       const float* __restrict__ w_fc,  // [10,131072] c*1024+y*32+x
                       const float* __restrict__ b_fc,
                       float* __restrict__ out) {       // [B,10]
    __shared__ float red[256];
    __shared__ float logit[10];
    const int b = blockIdx.x;
    const int t = threadIdx.x;
    float part[10];
    #pragma unroll
    for (int o = 0; o < 10; ++o) part[o] = 0.0f;
    for (int f = t; f < 131072; f += 256) {
        int c   = f >> 10;          // NCHW flat: f = c*1024 + y*32 + x
        int rem = f & 1023;
        float hv = h2[(b * 1024 + rem) * 128 + c];
        #pragma unroll
        for (int o = 0; o < 10; ++o) part[o] += hv * w_fc[o * 131072 + f];
    }
    for (int o = 0; o < 10; ++o) {
        red[t] = part[o];
        __syncthreads();
        for (int s = 128; s > 0; s >>= 1) {
            if (t < s) red[t] += red[t + s];
            __syncthreads();
        }
        if (t == 0) logit[o] = red[0] + b_fc[o];
        __syncthreads();
    }
    if (t == 0) {
        float mx = logit[0];
        for (int o = 1; o < 10; ++o) mx = fmaxf(mx, logit[o]);
        float se = 0.0f, e[10];
        for (int o = 0; o < 10; ++o) { e[o] = __expf(logit[o] - mx); se += e[o]; }
        for (int o = 0; o < 10; ++o) out[b * 10 + o] = e[o] / se;
    }
}

// ---------------------------------------------------------------------------
extern "C" void kernel_launch(void* const* d_in, const int* in_sizes, int n_in,
                              void* d_out, int out_size, void* d_ws, size_t ws_size,
                              hipStream_t stream) {
    const float* x     = (const float*)d_in[0];
    const float* w_in  = (const float*)d_in[1];
    const float* b_in  = (const float*)d_in[2];
    const float* w_rec = (const float*)d_in[3];
    const float* b_rec = (const float*)d_in[4];
    const float* w_c1  = (const float*)d_in[5];
    const float* b_c1  = (const float*)d_in[6];
    const float* w_c2  = (const float*)d_in[7];
    const float* b_c2  = (const float*)d_in[8];
    const float* w_fc  = (const float*)d_in[9];
    const float* b_fc  = (const float*)d_in[10];
    float* out = (float*)d_out;

    char*  ws  = (char*)d_ws;
    size_t off = 0;
    auto alloc = [&](size_t bytes) -> char* {
        char* p = ws + off;
        off = (off + bytes + 255) & ~(size_t)255;
        return p;
    };
    _Float16* spkA   = (_Float16*)alloc((size_t)NPIX * RCH * 2);
    _Float16* spkB   = (_Float16*)alloc((size_t)NPIX * RCH * 2);
    float*    syn    = (float*)   alloc((size_t)NPIX * RCH * 4);
    float*    mem    = (float*)   alloc((size_t)NPIX * RCH * 4);
    float*    cur    = (float*)   alloc((size_t)NPIX * RCH * 4);
    float*    psum   = (float*)   alloc((size_t)3 * NPIX * RCH * 4);
    _Float16* ro     = (_Float16*)alloc((size_t)NPIX * 192 * 2);
    _Float16* h1     = (_Float16*)alloc((size_t)NPIX * 256 * 2);
    float*    h2     = (float*)   alloc((size_t)NPIX * 128 * 4);
    _Float16* wrecpk = (_Float16*)alloc((size_t)64 * 64 * 25 * 2);
    _Float16* wc1pk  = (_Float16*)alloc((size_t)256 * 192 * 9 * 2);
    _Float16* wc2pk  = (_Float16*)alloc((size_t)128 * 256 * 9 * 2);

    // Zero-init state and partition accumulators (spkB is written before read).
    hipMemsetAsync(spkA, 0, (size_t)NPIX * RCH * 2, stream);
    hipMemsetAsync(syn,  0, (size_t)NPIX * RCH * 4, stream);
    hipMemsetAsync(mem,  0, (size_t)NPIX * RCH * 4, stream);
    hipMemsetAsync(psum, 0, (size_t)3 * NPIX * RCH * 4, stream);

    // Repack conv weights into WMMA B-fragment layout.
    pack_weights_kernel<<<(64 * 64 * 25 + 255) / 256, 256, 0, stream>>>(
        w_rec, wrecpk, 64, 64, 25);
    pack_weights_kernel<<<(256 * 192 * 9 + 255) / 256, 256, 0, stream>>>(
        w_c1, wc1pk, 256, 192, 9);
    pack_weights_kernel<<<(128 * 256 * 9 + 255) / 256, 256, 0, stream>>>(
        w_c2, wc2pk, 128, 256, 9);

    // Sequential LIF recurrence (inherent T dependence).
    const size_t step_elems = (size_t)BATCH * RCH * HDIM * WDIM;  // 2097152
    for (int t = 0; t < T_STEPS; ++t) {
        const float* xt = x + (size_t)t * BATCH * 2 * HDIM * WDIM;
        input_cur_kernel<<<(NPIX * RCH) / 256, 256, 0, stream>>>(xt, w_in, b_in, cur);
        const _Float16* sin  = (t & 1) ? spkB : spkA;
        _Float16*       sout = (t & 1) ? spkA : spkB;
        rsnn_step_kernel<<<1024, 256, 0, stream>>>(
            sin, cur, wrecpk, b_rec, syn, mem, sout,
            out + 320 + (size_t)t * step_elems,
            psum + (size_t)(t / 8) * NPIX * RCH);
    }

    // Readout head.
    build_readout_kernel<<<(NPIX * 192) / 256, 256, 0, stream>>>(psum, ro);
    conv_wmma_kernel<192, 256, 3, true ><<<(2048 * 16) / 8, 256, 0, stream>>>(
        ro, wc1pk, b_c1, h1, nullptr);
    conv_wmma_kernel<256, 128, 3, false><<<(2048 * 8) / 8, 256, 0, stream>>>(
        h1, wc2pk, b_c2, nullptr, h2);
    fc_softmax_kernel<<<BATCH, 256, 0, stream>>>(h2, w_fc, b_fc, out);
}
